// TwoLayerSparseConvModel_20564303414102
// MI455X (gfx1250) — matrix operand order repaired and checked by reading, not compiled, not run on hardware
//
#include <hip/hip_runtime.h>
#include <hip/hip_bf16.h>
#include <stddef.h>

// ---------------------------------------------------------------------------
// Problem constants (from the reference)
// ---------------------------------------------------------------------------
constexpr int NV = 262144;   // active voxels
constexpr int CC = 64;       // channels
constexpr int KK = 27;       // kernel offsets

// ---------------------------------------------------------------------------
// Feature detection (compile-safe: fall back to sync staging + __syncthreads)
// ---------------------------------------------------------------------------
#if __has_builtin(__builtin_amdgcn_global_load_async_to_lds_b128) && \
    __has_builtin(__builtin_amdgcn_s_wait_asynccnt)
#define HAVE_ASYNC 1
#else
#define HAVE_ASYNC 0
#endif

#if __has_builtin(__builtin_amdgcn_s_barrier_signal) && \
    __has_builtin(__builtin_amdgcn_s_barrier_wait) &&   \
    (HAVE_ASYNC || __has_builtin(__builtin_amdgcn_s_wait_dscnt))
#define HAVE_SPLIT_BAR 1
#else
#define HAVE_SPLIT_BAR 0
#endif

#define AS1 __attribute__((address_space(1)))
#define AS3 __attribute__((address_space(3)))

// ---------------------------------------------------------------------------
// Types for WMMA (CDNA5 / gfx1250, wave32)
// ---------------------------------------------------------------------------
typedef __bf16 bf16_t;
typedef __attribute__((ext_vector_type(16))) __bf16    v16bf;
typedef __attribute__((ext_vector_type(8)))  float     v8f;
typedef __attribute__((ext_vector_type(4)))  unsigned  v4u;
typedef __attribute__((ext_vector_type(4)))  int       v4i;

union Frag {            // one 16x32 (A) or 32x16 (B) bf16 fragment: 32 bytes/lane
    v16bf v;
    v4u   u[2];
};

__device__ __forceinline__ bf16_t f32_to_bf16(float f) {
    union { float f; unsigned u; } c{f};
    unsigned u = c.u;
    u += 0x7FFFu + ((u >> 16) & 1u);        // round-to-nearest-even
    union { unsigned short s; bf16_t b; } o{(unsigned short)(u >> 16)};
    return o.b;
}

// ---------------------------------------------------------------------------
// features f32 -> bf16 (one-time; the bf16 table then lives in L2)
// ---------------------------------------------------------------------------
__global__ __launch_bounds__(256) void cvt_feat(const float* __restrict__ x,
                                                bf16_t* __restrict__ y, int n) {
    int i = (blockIdx.x * 256 + threadIdx.x) * 4;
    if (i >= n) return;
#pragma unroll
    for (int j = 0; j < 4; ++j) y[i + j] = f32_to_bf16(x[i + j]);
}

// Zero the sentinel row (index NV) of both gather tables: masked-out lanes
// redirect their gather here instead of taking a divergent branch.
__global__ void zero_pad_rows(bf16_t* __restrict__ featb, bf16_t* __restrict__ hb) {
    int t = threadIdx.x;               // 64 threads
    featb[(size_t)NV * CC + t] = (bf16_t)0.0f;
    hb[(size_t)NV * CC + t]    = (bf16_t)0.0f;
}

// ---------------------------------------------------------------------------
// Pack weights w[k][c_in][c_out] (f32) into this file's contiguous-k per-lane layout of a
// 16-bit B-matrix 32x16 fragment (wave32):
//   lanes 0-15  hold K = 0..15  (VGPR j -> K = 2j, 2j+1), column N = lane
//   lanes 16-31 hold K = 16..31,                          column N = lane-16
// Fragment order: [k][nt][kfrag], each fragment = 512 bf16, lane-major 32 B.
// ---------------------------------------------------------------------------
__global__ __launch_bounds__(256) void pack_w(const float* __restrict__ w,
                                              bf16_t* __restrict__ out) {
    int t = blockIdx.x * 256 + threadIdx.x;       // 0 .. 27*8*512-1
    if (t >= KK * 8 * 512) return;
    int e  = t & 511;          // element within fragment
    int fb = t >> 9;           // fragment id = k*8 + nt*2 + f
    int f  = fb & 1;
    int nt = (fb >> 1) & 3;
    int k  = fb >> 3;
    int l  = e >> 4;           // lane
    int j  = (e >> 1) & 7;     // vgpr within fragment
    int h  = e & 1;            // low/high half of the 32-bit vgpr
    int kd = f * 32 + ((l >= 16) ? 16 : 0) + 2 * j + h;  // contraction (c_in)
    int n  = nt * 16 + (l & 15);                         // output channel
    out[t] = f32_to_bf16(w[((size_t)k * CC + kd) * CC + n]);
}

// ---------------------------------------------------------------------------
// Gather helpers
// ---------------------------------------------------------------------------
__device__ __forceinline__ int gather_idx(const int* __restrict__ nbr,
                                          const unsigned char* __restrict__ msk,
                                          int k, int mrow) {
    const size_t ko = (size_t)k * NV + mrow;
    const int    i  = nbr[ko];
    return msk[ko] ? i : NV;   // NV = zero sentinel row
}

__device__ __forceinline__ void load_a(const bf16_t* __restrict__ src, int idx,
                                       int sel, Frag& a0, Frag& a1) {
    // Row = idx; 8 chunks of 8 bf16 (16 B) per row.
    // frag0 lanes<16: chunks {0,1}; lanes>=16: {2,3}. frag1: +4.
    const v4u* rowp = (const v4u*)(src + (size_t)idx * CC);
    a0.u[0] = rowp[0 + 2 * sel];
    a0.u[1] = rowp[1 + 2 * sel];
    a1.u[0] = rowp[4 + 2 * sel];
    a1.u[1] = rowp[5 + 2 * sel];
}

// ---------------------------------------------------------------------------
// One sparse-conv layer. One wave per 16-row M-tile, 8 waves per block.
// B fragments double-buffered through LDS (8 KB/offset, shared by all waves,
// staged by async global->LDS copies when available); A gathers and idx/mask
// loads software-pipelined 1 and 2 offsets ahead. k-loop unrolled by 2 so the
// A double-buffers swap statically (no v_mov rotation, no WMMA hazard nops).
// ---------------------------------------------------------------------------
template <bool OUT_BF16>
__global__ __launch_bounds__(256) void sconv(
    const bf16_t* __restrict__ src,        // gather source [NV+1][64] bf16
    const int* __restrict__ nbr,           // [KK][NV]
    const unsigned char* __restrict__ msk, // [KK][NV] (numpy bool)
    const bf16_t* __restrict__ wp,         // packed weights [KK][8][512]
    const float* __restrict__ bias,        // [64]
    const float* __restrict__ resid,       // [NV][64] f32 (layer 2 only)
    float* __restrict__ outf,              // f32 output (layer 2)
    bf16_t* __restrict__ outh)             // bf16 output (layer 1)
{
    __shared__ bf16_t ldsw[2][8 * 512];    // double-buffered B for one offset

    const int tid   = threadIdx.x;
    const int lane  = tid & 31;
    const int wave  = tid >> 5;
    const int mbase = (blockIdx.x * 8 + wave) * 16;
    const int mrow  = mbase + (lane & 15);   // row this lane gathers for A
    const int sel   = lane >> 4;             // lane-half -> A chunk pair

    const v8f zacc = {0.f, 0.f, 0.f, 0.f, 0.f, 0.f, 0.f, 0.f};
    v8f acc[4];
#pragma unroll
    for (int nt = 0; nt < 4; ++nt) acc[nt] = zacc;

    // Stage B[k] (8 KB) into LDS buffer `buf`, cooperatively (32 B/thread).
    auto stage_b = [&](int k, int buf) {
#if HAVE_ASYNC
        const char* g = (const char*)(wp + (size_t)k * 4096) + tid * 16;
        char*       l = (char*)&ldsw[buf][0] + tid * 16;
        __builtin_amdgcn_global_load_async_to_lds_b128(
            (AS1 v4i*)(g),        (AS3 v4i*)(l),        0, 0);
        __builtin_amdgcn_global_load_async_to_lds_b128(
            (AS1 v4i*)(g + 4096), (AS3 v4i*)(l + 4096), 0, 0);
#else
        const v4u* g = (const v4u*)(wp + (size_t)k * 4096);
        v4u*       s = (v4u*)&ldsw[buf][0];
        s[tid]       = g[tid];
        s[tid + 256] = g[tid + 256];
#endif
    };

    // Barrier that only drains what LDS visibility needs; in-flight L2 gathers
    // (LOADcnt) stay pending across it.
    auto barrier = [&]() {
#if HAVE_SPLIT_BAR
        asm volatile("" ::: "memory");
#if HAVE_ASYNC
        __builtin_amdgcn_s_wait_asynccnt(0);
#else
        __builtin_amdgcn_s_wait_dscnt(0);
#endif
        __builtin_amdgcn_s_barrier_signal(-1);
        __builtin_amdgcn_s_barrier_wait(-1);
        asm volatile("" ::: "memory");
#else
        __syncthreads();
#endif
    };

    // ---- prologue: B[0] -> LDS buf0, A[0] -> regs, idx[1] prefetched
    stage_b(0, 0);
    Frag A0, A1, B0, B1;
    {
        const int i0 = gather_idx(nbr, msk, 0, mrow);
        load_a(src, i0, sel, A0, A1);
    }
    int idx_nxt = gather_idx(nbr, msk, 1, mrow);
    barrier();

    // One pipeline stage: consume (ca0,ca1) + LDS buf (k&1); prefetch into
    // (na0,na1) and the other LDS buf.
    auto step = [&](int k, Frag& ca0, Frag& ca1, Frag& na0, Frag& na1,
                    bool last) {
        const int cur = k & 1;

        if (!last) {
            stage_b(k + 1, cur ^ 1);             // B[k+1] -> other LDS buf
            load_a(src, idx_nxt, sel, na0, na1); // A[k+1] gather (L2)
        }
        if (k + 2 < KK) idx_nxt = gather_idx(nbr, msk, k + 2, mrow);

        // 8 WMMAs: 4 independent accumulation chains, B streamed from LDS.
        const v4u* lb = (const v4u*)&ldsw[cur][0];
#pragma unroll
        for (int nt = 0; nt < 4; ++nt) {
            Frag b0, b1;
            const v4u* p0 = lb + (nt * 2 + 0) * 64 + lane * 2;
            const v4u* p1 = lb + (nt * 2 + 1) * 64 + lane * 2;
            b0.u[0] = p0[0]; b0.u[1] = p0[1];
            b1.u[0] = p1[0]; b1.u[1] = p1[1];
            acc[nt] = __builtin_amdgcn_wmma_f32_16x16x32_bf16(
                false, ca0.v, false, b0.v, (short)0, acc[nt], false, false);
            acc[nt] = __builtin_amdgcn_wmma_f32_16x16x32_bf16(
                false, ca1.v, false, b1.v, (short)0, acc[nt], false, false);
        }

        if (!last) barrier();
    };

    // KK = 27: 13 unrolled pairs (0..25), then the final offset 26.
    for (int k = 0; k + 2 < KK; k += 2) {
        step(k,     A0, A1, B0, B1, false);
        step(k + 1, B0, B1, A0, A1, false);
    }
    step(KK - 1, A0, A1, B0, B1, true);

    // Epilogue. D layout: VGPR r -> row = mbase + r + sel*8, col = nt*16+(lane&15)
    const int colb = lane & 15;
    const int rofs = sel * 8;
#pragma unroll
    for (int nt = 0; nt < 4; ++nt) {
        const int   col = nt * 16 + colb;
        const float bv  = bias[col];
#pragma unroll
        for (int r = 0; r < 8; ++r) {
            const size_t pos = (size_t)(mbase + rofs + r) * CC + col;
            const float  v   = acc[nt][r] + bv;
            if (OUT_BF16) outh[pos] = f32_to_bf16(v);
            else          outf[pos] = v + resid[pos];
        }
    }
}

// ---------------------------------------------------------------------------
// Host-side launch. Workspace layout (256 B aligned):
//   featb : (NV+1) x 64 bf16   (sentinel zero row at NV)
//   hb    : (NV+1) x 64 bf16
//   w1p   : packed weights (221184 B, padded)
//   w2p   : packed weights
// ---------------------------------------------------------------------------
extern "C" void kernel_launch(void* const* d_in, const int* in_sizes, int n_in,
                              void* d_out, int out_size, void* d_ws, size_t ws_size,
                              hipStream_t stream) {
    const float*         feat = (const float*)d_in[0];
    const int*           nbr  = (const int*)d_in[1];
    const unsigned char* msk  = (const unsigned char*)d_in[2];
    const float*         w1   = (const float*)d_in[3];
    const float*         b1   = (const float*)d_in[4];
    const float*         w2   = (const float*)d_in[5];
    const float*         b2   = (const float*)d_in[6];
    float*               out  = (float*)d_out;

    char* ws = (char*)d_ws;
    const size_t FEAT_B = (((size_t)(NV + 1) * CC * sizeof(bf16_t)) + 255) & ~(size_t)255;
    bf16_t* featb = (bf16_t*)(ws);
    bf16_t* hb    = (bf16_t*)(ws + FEAT_B);
    bf16_t* w1p   = (bf16_t*)(ws + 2 * FEAT_B);
    bf16_t* w2p   = (bf16_t*)(ws + 2 * FEAT_B + 262144);

    const int nfeat = NV * CC;
    cvt_feat<<<nfeat / (256 * 4), 256, 0, stream>>>(feat, featb, nfeat);
    zero_pad_rows<<<1, 64, 0, stream>>>(featb, hb);

    const int npack = KK * 8 * 512;
    pack_w<<<(npack + 255) / 256, 256, 0, stream>>>(w1, w1p);
    pack_w<<<(npack + 255) / 256, 256, 0, stream>>>(w2, w2p);

    const int nblocks = NV / (16 * 8);   // one wave per 16-row tile, 8 waves/block
    sconv<true ><<<nblocks, 256, 0, stream>>>(featb, nbr, msk, w1p, b1,
                                              nullptr, nullptr, hb);
    sconv<false><<<nblocks, 256, 0, stream>>>(hb, nbr, msk, w2p, b2,
                                              feat, out, nullptr);
}
